// PathConvBlock_14164802142827
// MI455X (gfx1250) — matrix operand order repaired and checked by reading, not compiled, not run on hardware
//
#include <hip/hip_runtime.h>

// ---------------- problem constants ----------------
#define BATCH 32
#define DIMC  384
#define LLEN  2048
#define HID   1536            // EXP*DIM
#define COLS  (BATCH*LLEN)    // 65536 GEMM columns
#define NT    32              // columns per workgroup in fused GEMM

typedef unsigned int uint;
typedef __attribute__((ext_vector_type(16))) __bf16 v16bf;
typedef __attribute__((ext_vector_type(8)))  float  v8f;
typedef __attribute__((ext_vector_type(4)))  uint   u32x4;
typedef __attribute__((ext_vector_type(8)))  int    i32x8;
typedef __attribute__((ext_vector_type(4)))  int    i32x4;

union Frag16 { v16bf v; uint4 q[2]; };

#if defined(__AMDGCN__) && __has_builtin(__builtin_amdgcn_tensor_load_to_lds)
#define HAVE_TDM 1
#else
#define HAVE_TDM 0
#endif

__device__ __forceinline__ unsigned short f32_to_bf16(float f) {
    uint u = __float_as_uint(f);
    u += 0x7FFFu + ((u >> 16) & 1u);          // round-to-nearest-even
    return (unsigned short)(u >> 16);
}

__device__ __forceinline__ v8f wmma_bf16(const Frag16& a, const Frag16& b, v8f c) {
    return __builtin_amdgcn_wmma_f32_16x16x32_bf16(false, a.v, false, b.v,
                                                   (short)0, c, false, false);
}

// ---------------- K0: weight f32 -> bf16 ----------------
__global__ __launch_bounds__(256)
void cvt_weights_kernel(const float* __restrict__ w1, const float* __restrict__ w2,
                        unsigned short* __restrict__ w1b, unsigned short* __restrict__ w2b,
                        int n) {
    int i = blockIdx.x * 256 + threadIdx.x;
    if (i < n) { w1b[i] = f32_to_bf16(w1[i]); w2b[i] = f32_to_bf16(w2[i]); }
}

// ---------------- K1: EnhancedPathAwareECA -> per-(b,c) scale = 1 + attn*w ----------------
__global__ __launch_bounds__(384)
void eca_kernel(const float* __restrict__ x,
                const float* __restrict__ ew1, const float* __restrict__ ew2,
                const float* __restrict__ cw,  const float* __restrict__ cb,
                const float* __restrict__ lnw, const float* __restrict__ lnb,
                const float* __restrict__ mw1, const float* __restrict__ mb1,
                const float* __restrict__ mw2, const float* __restrict__ mb2,
                float* __restrict__ scaleOut) {
    __shared__ float yS[384];
    __shared__ float attnS[384];
    __shared__ float featS[8];
    __shared__ float wS[8];
    const int b = blockIdx.x;
    const int tid = threadIdx.x, lane = tid & 31, wave = tid >> 5;
    const float* xb = x + (size_t)b * DIMC * LLEN;

    // channel means over L (one wave per channel, coalesced)
    for (int c = wave; c < DIMC; c += 12) {
        const float* row = xb + (size_t)c * LLEN;
        float s = 0.f;
        for (int j = lane; j < LLEN; j += 32) s += row[j];
        for (int off = 16; off; off >>= 1) s += __shfl_xor(s, off, 32);
        if (lane == 0) yS[c] = s * (1.f / LLEN);
    }
    __syncthreads();

    // per-path 1D convs along D (zero pad inside each 48-wide path), combine, sigmoid
    {
        const int p = tid / 48, d = tid % 48;
        float y1 = 0.f, y2 = 0.f;
        #pragma unroll
        for (int k = 0; k < 3; ++k) { int dd = d + k - 1; if (dd >= 0 && dd < 48) y1 += yS[p*48+dd] * ew1[p*3+k]; }
        #pragma unroll
        for (int k = 0; k < 5; ++k) { int dd = d + k - 2; if (dd >= 0 && dd < 48) y2 += yS[p*48+dd] * ew2[p*5+k]; }
        float yc = cw[0] * y1 + cw[1] * y2 + cb[0];
        attnS[tid] = 1.f / (1.f + expf(-yc));
    }
    __syncthreads();
    if (tid < 8) {
        float s = 0.f;
        for (int d = 0; d < 48; ++d) s += attnS[tid*48 + d];
        featS[tid] = s * (1.f / 48.f);
    }
    __syncthreads();
    if (tid == 0) {  // tiny 8-wide LN + MLP, serial
        float mu = 0.f; for (int i = 0; i < 8; ++i) mu += featS[i]; mu *= 0.125f;
        float var = 0.f; for (int i = 0; i < 8; ++i) { float d = featS[i]-mu; var += d*d; } var *= 0.125f;
        float rs = rsqrtf(var + 1e-5f);
        float nrm[8], h[16];
        for (int i = 0; i < 8; ++i) nrm[i] = (featS[i]-mu) * rs * lnw[i] + lnb[i];
        for (int o = 0; o < 16; ++o) {
            float a = mb1[o];
            for (int i = 0; i < 8; ++i) a += nrm[i] * mw1[o*8+i];
            h[o] = 0.5f * a * (1.f + erff(a * 0.70710678f));
        }
        for (int p = 0; p < 8; ++p) {
            float a = mb2[p];
            for (int o = 0; o < 16; ++o) a += h[o] * mw2[p*16+o];
            wS[p] = 1.f / (1.f + expf(-a));
        }
    }
    __syncthreads();
    scaleOut[(size_t)b * DIMC + tid] = 1.f + attnS[tid] * wS[tid / 48];
}

// ---------------- K2: x2 = x*scale -> depthwise conv k=9 -> channel-first LN -> bf16 [col][c] ----------------
__global__ __launch_bounds__(384)
void dwconv_ln_kernel(const float* __restrict__ x, const float* __restrict__ scale,
                      const float* __restrict__ dww, const float* __restrict__ dwb,
                      const float* __restrict__ lnw, const float* __restrict__ lnb,
                      unsigned short* __restrict__ outT) {
    __shared__ float red[24];
    __shared__ float stat[2];
    const int col = blockIdx.x;            // = b*2048 + l
    const int b = col >> 11, l = col & 2047;
    const int c = threadIdx.x;
    const float s = scale[b * DIMC + c];
    const float* xr = x + ((size_t)b * DIMC + c) * LLEN;
    float acc = 0.f;
    #pragma unroll
    for (int k = 0; k < 9; ++k) {
        int ll = l + k - 4;
        float xv = (ll >= 0 && ll < LLEN) ? xr[ll] : 0.f;
        acc += xv * dww[c * 9 + k];
    }
    acc = acc * s + dwb[c];                // per-channel scale commutes with conv

    float v = acc, v2 = acc * acc;
    for (int off = 16; off; off >>= 1) { v += __shfl_xor(v, off, 32); v2 += __shfl_xor(v2, off, 32); }
    const int wave = threadIdx.x >> 5, lane = threadIdx.x & 31;
    if (lane == 0) { red[wave] = v; red[12 + wave] = v2; }
    __syncthreads();
    if (threadIdx.x == 0) {
        float s1 = 0.f, s2 = 0.f;
        for (int i = 0; i < 12; ++i) { s1 += red[i]; s2 += red[12 + i]; }
        float mu = s1 * (1.f / DIMC);
        stat[0] = mu;
        stat[1] = rsqrtf(s2 * (1.f / DIMC) - mu * mu + 1e-6f);
    }
    __syncthreads();
    float nv = lnw[c] * (acc - stat[0]) * stat[1] + lnb[c];
    outT[(size_t)col * DIMC + c] = f32_to_bf16(nv);
}

// ---------------- K3: fused  W1@B +b1 -> LN -> GELU -> W2@hh +b2 -> *scaling + x ----------------
// LDS: hhF f32 [32][1540] (197120 B) | union{ Bsh TDM-padded 32x816B, hhB bf16 [32][1544] } (98816 B) | stats
// B tile LDS layout (TDM pad: +16B per 256B of stream): row n at n*816, byte e at e + 16*(e>>8).
#define HHF_STRIDE 1540
#define BSH_ROWB   816
#define HHB_STRIDE 1544
#define LDS_HHF_BYTES (NT * HHF_STRIDE * 4)
#define LDS_UNI_BYTES (NT * HHB_STRIDE * 2)

__global__ __launch_bounds__(256)
void fused_mlp_kernel(const unsigned short* __restrict__ w1b,   // [1536][384] bf16
                      const unsigned short* __restrict__ w2b,   // [384][1536] bf16
                      const unsigned short* __restrict__ outT,  // [COLS][384] bf16
                      const float* __restrict__ ib_b1,
                      const float* __restrict__ ibln_w, const float* __restrict__ ibln_b,
                      const float* __restrict__ ib_b2, const float* __restrict__ scaling,
                      const float* __restrict__ x,
                      float* __restrict__ out) {
    extern __shared__ char smem[];
    float*          hhF  = (float*)smem;
    char*           BshB = smem + LDS_HHF_BYTES;                 // padded B tile (bytes)
    unsigned short* hhB  = (unsigned short*)(smem + LDS_HHF_BYTES);
    float*          stat = (float*)(smem + LDS_HHF_BYTES + LDS_UNI_BYTES);

    const int tid  = threadIdx.x;
    const int lane = tid & 31;
    const int wave = tid >> 5;
    const int half = lane >> 4;       // selects K sub-range per WMMA 16-bit layout
    const int ln16 = lane & 15;
    const int colBase = blockIdx.x * NT;
    const unsigned short* gsrc = outT + (size_t)colBase * DIMC;

    // ---- stage B tile: 32 rows x 768B, TDM-padded to 816B/row ----
#if HAVE_TDM
    if (wave == 0) {
        unsigned long long ga = (unsigned long long)(size_t)gsrc;
        u32x4 g0;
        g0[0] = 1u;                                            // count=1, load, user mode
        g0[1] = (uint)(size_t)(void*)BshB;                     // lds_addr (low 32 bits of flat)
        g0[2] = (uint)(ga & 0xFFFFFFFFu);
        g0[3] = (uint)((ga >> 32) & 0x1FFFFFFu) | (2u << 30);  // global_addr hi | type=2
        i32x8 g1;
        g1[0] = (1 << 16)            // data_size = 1 (2 bytes)
              | (1 << 20)            // pad_enable
              | (5 << 22)            // pad_interval code 5 -> every 256B
              | (3 << 25);           // pad_amount  code 3 -> 4 DWORDs = 16B
        g1[1] = (DIMC & 0xFFFF) << 16;          // tensor_dim0 lo16 (atomic barrier off)
        g1[2] = (DIMC >> 16) | (NT << 16);      // tensor_dim0 hi16 | tensor_dim1 lo16
        g1[3] = (DIMC << 16);                   // tensor_dim1 hi16 | tile_dim0 = 384
        g1[4] = NT;                             // tile_dim1 = 32, tile_dim2 = 0
        g1[5] = DIMC;                           // tensor_dim0_stride lo32 = 384 elements
        g1[6] = 0;
        g1[7] = 0;
        i32x4 z4 = {0, 0, 0, 0};
#if defined(__clang_major__) && __clang_major__ >= 23
        i32x8 z8 = {0, 0, 0, 0, 0, 0, 0, 0};
        __builtin_amdgcn_tensor_load_to_lds(g0, g1, z4, z4, z8, 0);
#else
        __builtin_amdgcn_tensor_load_to_lds(g0, g1, z4, z4, 0);
#endif
        __builtin_amdgcn_s_wait_tensorcnt(0);
    }
#else
    // manual fallback producing the identical padded layout
    for (int i = tid; i < NT * 48; i += 256) {          // 48 uint4 per 768B row
        int n = i / 48, q = i % 48;
        uint4 d = ((const uint4*)gsrc)[(size_t)n * 48 + q];
        *(uint4*)(BshB + n * BSH_ROWB + q * 16 + 16 * (q / 16)) = d;
    }
#endif
    __syncthreads();

    // ---- GEMM1: hh[1536 x 32]; 2x2 register-blocked 16x16 tiles, K=384 ----
    for (int t = wave; t < 48; t += 8) {
        const int m0 = t * 32;                          // two M tiles: m0, m0+16
        v8f a00 = {0.f,0.f,0.f,0.f,0.f,0.f,0.f,0.f}, a01 = a00, a10 = a00, a11 = a00;
        const unsigned short* ar0 = w1b + (size_t)(m0 + ln16) * DIMC + half * 16;
        const unsigned short* ar1 = ar0 + 16 * DIMC;
        const char* bc0 = BshB + ln16 * BSH_ROWB;        // nt=0 column
        const char* bc1 = BshB + (16 + ln16) * BSH_ROWB; // nt=1 column
        #pragma unroll
        for (int k0 = 0; k0 < DIMC; k0 += 32) {
            Frag16 fa0, fa1, fb0, fb1;
            __builtin_prefetch(ar0 + k0 + 128, 0, 1);
            fa0.q[0] = *(const uint4*)(ar0 + k0);
            fa0.q[1] = *(const uint4*)(ar0 + k0 + 8);
            fa1.q[0] = *(const uint4*)(ar1 + k0);
            fa1.q[1] = *(const uint4*)(ar1 + k0 + 8);
            const unsigned eb  = 2u * (unsigned)k0 + 32u * (unsigned)half;
            const unsigned off = eb + 16u * (eb >> 8);   // TDM pad mapping (folds: full unroll)
            fb0.q[0] = *(const uint4*)(bc0 + off);
            fb0.q[1] = *(const uint4*)(bc0 + off + 16);
            fb1.q[0] = *(const uint4*)(bc1 + off);
            fb1.q[1] = *(const uint4*)(bc1 + off + 16);
            a00 = wmma_bf16(fa0, fb0, a00);
            a01 = wmma_bf16(fa0, fb1, a01);
            a10 = wmma_bf16(fa1, fb0, a10);
            a11 = wmma_bf16(fa1, fb1, a11);
        }
        const int mbA = m0 + half * 8, mbB = m0 + 16 + half * 8;
        float* d00 = hhF + ln16 * HHF_STRIDE + mbA;
        float* d10 = hhF + ln16 * HHF_STRIDE + mbB;
        float* d01 = hhF + (16 + ln16) * HHF_STRIDE + mbA;
        float* d11 = hhF + (16 + ln16) * HHF_STRIDE + mbB;
        #pragma unroll
        for (int r = 0; r < 8; ++r) {
            const float bA = ib_b1[mbA + r], bB = ib_b1[mbB + r];
            d00[r] = a00[r] + bA;  d01[r] = a01[r] + bA;
            d10[r] = a10[r] + bB;  d11[r] = a11[r] + bB;
        }
    }
    __syncthreads();

    // ---- column LN stats over 1536 (one wave per 4 columns) ----
    for (int n = wave * 4; n < wave * 4 + 4; ++n) {
        float s = 0.f, s2 = 0.f;
        const float* cp = hhF + n * HHF_STRIDE;
        for (int k = lane; k < HID; k += 32) { float v = cp[k]; s += v; s2 += v * v; }
        for (int off = 16; off; off >>= 1) { s += __shfl_xor(s, off, 32); s2 += __shfl_xor(s2, off, 32); }
        if (lane == 0) {
            float mu = s * (1.f / HID);
            stat[n * 2]     = mu;
            stat[n * 2 + 1] = rsqrtf(s2 * (1.f / HID) - mu * mu + 1e-6f);
        }
    }
    __syncthreads();

    // ---- LN + GELU + bf16 pack (overwrites B region, no longer needed) ----
    for (int i = tid; i < NT * HID; i += 256) {
        int n = i / HID, k = i % HID;
        float v  = hhF[n * HHF_STRIDE + k];
        float nv = ibln_w[k] * (v - stat[n * 2]) * stat[n * 2 + 1] + ibln_b[k];
        float g  = 0.5f * nv * (1.f + erff(nv * 0.70710678f));
        hhB[n * HHB_STRIDE + k] = f32_to_bf16(g);
    }
    __syncthreads();

    // ---- GEMM2: [384 x 32]; 3x2 register-blocked (one macro-tile per wave), K=1536 ----
    {
        const int m0 = wave * 48;                       // 8 waves * 48 rows = 384
        v8f acc[6];
        #pragma unroll
        for (int i = 0; i < 6; ++i) acc[i] = (v8f){0.f,0.f,0.f,0.f,0.f,0.f,0.f,0.f};
        const unsigned short* ar0 = w2b + (size_t)(m0      + ln16) * HID + half * 8;
        const unsigned short* ar1 = w2b + (size_t)(m0 + 16 + ln16) * HID + half * 8;
        const unsigned short* ar2 = w2b + (size_t)(m0 + 32 + ln16) * HID + half * 8;
        const unsigned short* bc0 = hhB + ln16        * HHB_STRIDE + half * 8;
        const unsigned short* bc1 = hhB + (16 + ln16) * HHB_STRIDE + half * 8;
        #pragma unroll 4
        for (int k0 = 0; k0 < HID; k0 += 32) {
            Frag16 f0, f1, f2, g0, g1;
            __builtin_prefetch(ar0 + k0 + 128, 0, 1);
            f0.q[0] = *(const uint4*)(ar0 + k0); f0.q[1] = *(const uint4*)(ar0 + k0 + 16);
            f1.q[0] = *(const uint4*)(ar1 + k0); f1.q[1] = *(const uint4*)(ar1 + k0 + 16);
            f2.q[0] = *(const uint4*)(ar2 + k0); f2.q[1] = *(const uint4*)(ar2 + k0 + 16);
            g0.q[0] = *(const uint4*)(bc0 + k0); g0.q[1] = *(const uint4*)(bc0 + k0 + 16);
            g1.q[0] = *(const uint4*)(bc1 + k0); g1.q[1] = *(const uint4*)(bc1 + k0 + 16);
            acc[0] = wmma_bf16(f0, g0, acc[0]);
            acc[1] = wmma_bf16(f0, g1, acc[1]);
            acc[2] = wmma_bf16(f1, g0, acc[2]);
            acc[3] = wmma_bf16(f1, g1, acc[3]);
            acc[4] = wmma_bf16(f2, g0, acc[4]);
            acc[5] = wmma_bf16(f2, g1, acc[5]);
        }
        #pragma unroll
        for (int ms = 0; ms < 3; ++ms) {
            #pragma unroll
            for (int nt = 0; nt < 2; ++nt) {
                const v8f& a = acc[ms * 2 + nt];
                const int n   = nt * 16 + ln16;
                const int col = colBase + n;
                const int b   = col >> 11, l = col & 2047;
                const int cb  = m0 + ms * 16 + half * 8;
                const size_t base = ((size_t)b * DIMC + cb) * LLEN + l;
                #pragma unroll
                for (int r = 0; r < 8; ++r) {
                    float v = (a[r] + ib_b2[cb + r]) * scaling[cb + r];
                    out[base + (size_t)r * LLEN] = v + x[base + (size_t)r * LLEN];
                }
            }
        }
    }
}

// ---------------- launch ----------------
extern "C" void kernel_launch(void* const* d_in, const int* in_sizes, int n_in,
                              void* d_out, int out_size, void* d_ws, size_t ws_size,
                              hipStream_t stream) {
    const float* x         = (const float*)d_in[0];
    const float* eca_w1    = (const float*)d_in[1];
    const float* eca_w2    = (const float*)d_in[2];
    const float* combine_w = (const float*)d_in[3];
    const float* combine_b = (const float*)d_in[4];
    const float* mlp_ln_w  = (const float*)d_in[5];
    const float* mlp_ln_b  = (const float*)d_in[6];
    const float* mlp_w1    = (const float*)d_in[7];
    const float* mlp_b1    = (const float*)d_in[8];
    const float* mlp_w2    = (const float*)d_in[9];
    const float* mlp_b2    = (const float*)d_in[10];
    const float* dw_w      = (const float*)d_in[11];
    const float* dw_b      = (const float*)d_in[12];
    const float* dwln_w    = (const float*)d_in[13];
    const float* dwln_b    = (const float*)d_in[14];
    const float* ib_w1     = (const float*)d_in[15];
    const float* ib_b1     = (const float*)d_in[16];
    const float* ibln_w    = (const float*)d_in[17];
    const float* ibln_b    = (const float*)d_in[18];
    const float* ib_w2     = (const float*)d_in[19];
    const float* ib_b2     = (const float*)d_in[20];
    const float* scaling   = (const float*)d_in[21];

    char* ws = (char*)d_ws;
    float*          scaleBuf = (float*)ws;                                  // 49152 B
    unsigned short* w1b      = (unsigned short*)(ws + 49152);               // 1179648 B
    unsigned short* w2b      = (unsigned short*)(ws + 49152 + 1179648);     // 1179648 B
    unsigned short* outT     = (unsigned short*)(ws + 49152 + 2 * 1179648); // 50331648 B

    const int NW = HID * DIMC;  // 589824
    cvt_weights_kernel<<<(NW + 255) / 256, 256, 0, stream>>>(ib_w1, ib_w2, w1b, w2b, NW);
    eca_kernel<<<BATCH, 384, 0, stream>>>(x, eca_w1, eca_w2, combine_w, combine_b,
                                          mlp_ln_w, mlp_ln_b, mlp_w1, mlp_b1, mlp_w2, mlp_b2,
                                          scaleBuf);
    dwconv_ln_kernel<<<COLS, 384, 0, stream>>>(x, scaleBuf, dw_w, dw_b, dwln_w, dwln_b, outT);

    const size_t lds = LDS_HHF_BYTES + LDS_UNI_BYTES + 256;   // 296192 B < 320 KB
    fused_mlp_kernel<<<COLS / NT, 256, lds, stream>>>(w1b, w2b, outT, ib_b1,
                                                      ibln_w, ibln_b, ib_b2, scaling,
                                                      x, (float*)d_out);
}
